// GraphAttention_60275571032148
// MI455X (gfx1250) — matrix operand order repaired and checked by reading, not compiled, not run on hardware
//
#include <hip/hip_runtime.h>
#include <hip/hip_bf16.h>

typedef __attribute__((ext_vector_type(16))) _Float16 v16h;
typedef __attribute__((ext_vector_type(8)))  float    v8f;
typedef __attribute__((ext_vector_type(2)))  float    v2f;

static constexpr int KF   = 128;   // feature / filter dim
static constexpr int NMAX = 2048;  // max graph nodes (LDS staging of s row)
static constexpr int PART = 4;     // waves per block == j-partitions

// ---------------------------------------------------------------------------
// Kernel 1: h = x @ W  (exact f32 WMMA 16x16x4);  h stored TRANSPOSED as f16
//           (hT[(b*KF+col)*N + j]) so kernel 2's WMMA B-operands are
//           pairs of contiguous 16B runs.  s = h @ a_w + a_b reduced with shfl.
// One wave (32 threads) per 16-row tile of the flattened [B*N, KF] matrix.
// ---------------------------------------------------------------------------
__global__ __launch_bounds__(32) void gat_proj_kernel(
    const float* __restrict__ x, const float* __restrict__ W,
    const float* __restrict__ a_w, const float* __restrict__ a_b,
    _Float16* __restrict__ hT, float* __restrict__ s, int N)
{
  const int lane = threadIdx.x & 31;
  const int n    = lane & 15;   // column within 16-wide tile / A-operand row
  const int hf   = lane >> 4;   // lane half
  const long r0  = (long)blockIdx.x * 16;
  const long b   = r0 / N;          // batch index (16 | N, never straddles)
  const long j0b = r0 - b * N;      // within-batch row of this tile

  // Preload x A-operand fragments for all 32 K-steps (K=4 each).
  // f32 A 16x4 layout: lanes 0-15 hold K=0,1 in VGPR0/1; lanes 16-31 hold K=2,3.
  v2f xa[32];
  const float* xrow = x + (r0 + n) * KF;
#pragma unroll
  for (int kk = 0; kk < 32; ++kk) {
    const int k0 = 4 * kk + 2 * hf;
    xa[kk].x = xrow[k0];
    xa[kk].y = xrow[k0 + 1];
  }

  float sacc[8];
#pragma unroll
  for (int r = 0; r < 8; ++r) sacc[r] = 0.f;

  for (int g = 0; g < 8; ++g) {
    const int col = g * 16 + n;
    v8f acc = {};
#pragma unroll
    for (int kk = 0; kk < 32; ++kk) {
      const int k0 = 4 * kk + 2 * hf;
      v2f bb;                       // f32 B 4x16: mirror of A lane-half K split
      bb.x = W[(long)k0 * KF + col];
      bb.y = W[(long)(k0 + 1) * KF + col];
      acc = __builtin_amdgcn_wmma_f32_16x16x4_f32(
          false, xa[kk], false, bb, (short)0, acc, false, false);
    }
    // C layout: VGPR r -> row (r + 8*half), col n.  Store transposed f16.
    const float aw = a_w[col];
    _Float16* hTcol = hT + (b * KF + col) * (long)N + j0b;
#pragma unroll
    for (int r = 0; r < 8; ++r) {
      hTcol[r + 8 * hf] = (_Float16)acc[r];
      sacc[r] += acc[r] * aw;
    }
  }
  // Reduce s partials across the 16 column-lanes of each half.
#pragma unroll
  for (int r = 0; r < 8; ++r) {
    float v = sacc[r];
    v += __shfl_xor(v, 1, 32);
    v += __shfl_xor(v, 2, 32);
    v += __shfl_xor(v, 4, 32);
    v += __shfl_xor(v, 8, 32);
    sacc[r] = v;
  }
  if (n == 0) {
    const float ab = a_b[0];
#pragma unroll
    for (int r = 0; r < 8; ++r)
      s[r0 + r + 8 * hf] = sacc[r] + ab;
  }
}

// ---------------------------------------------------------------------------
// Kernel 2: split-flash fused scores + softmax + (attn @ h).
// 4 waves per block share one 16-row tile; wave w owns j-chunks
// j0 = 32w + 128k (interleaved). Reads A exactly once (67 MB total).
// Online softmax in f32; probabilities in f16 feeding V_WMMA_F32_16X16X32_F16.
// Deterministic LDS merge of the 4 partial (m, l, acc) states at the end.
// ---------------------------------------------------------------------------
__global__ __launch_bounds__(32 * PART) void gat_attn_kernel(
    const float* __restrict__ A, const float* __restrict__ bias,
    const _Float16* __restrict__ hT, const float* __restrict__ s,
    float* __restrict__ out, int N)
{
  __shared__ float s_sh[NMAX];            //  8 KB: attention scalars s[b,:]
  __shared__ float part[PART * 16 * KF];  // 32 KB: per-wave scaled partials
  __shared__ float sm[PART][16];          // per-wave row maxima
  __shared__ float sl[PART][16];          // per-wave row exp-sums

  const int tid  = threadIdx.x;
  const int lane = tid & 31;
  const int w    = tid >> 5;             // wave id == j partition
  const int n    = lane & 15;
  const int hf   = lane >> 4;
  const int b    = blockIdx.y;
  const int i0   = blockIdx.x * 16;
  const long bN  = (long)b * N;

  for (int idx = tid; idx < N; idx += 32 * PART) s_sh[idx] = s[bN + idx];
  __syncthreads();

  const int   myrow = i0 + n;            // stats row tracked by this lane
  const float si    = s_sh[myrow];
  const float* Arow = A + (bN + myrow) * (long)N;
  const _Float16* hTb = hT + bN * KF;    // batch base of transposed h

  v8f acc[8];
#pragma unroll
  for (int g = 0; g < 8; ++g) acc[g] = (v8f){};
  float m_run = -__builtin_inff();
  float l_run = 0.f;

  for (int j0 = 32 * w; j0 < N; j0 += 32 * PART) {
    // f16 A-operand K map: K(t) = 16*(t>>3) + 8*half + (t&7)  (two 8-runs).
    const float* Ap = Arow + j0 + 8 * hf;
    const float* sp = s_sh + j0 + 8 * hf;
    float4 av[4], sv[4];
    av[0] = *(const float4*)(Ap);      av[1] = *(const float4*)(Ap + 4);
    av[2] = *(const float4*)(Ap + 16); av[3] = *(const float4*)(Ap + 20);
    sv[0] = *(const float4*)(sp);      sv[1] = *(const float4*)(sp + 4);
    sv[2] = *(const float4*)(sp + 16); sv[3] = *(const float4*)(sp + 20);
    __builtin_prefetch(Ap + 32 * PART, 0, 1);  // next chunk for this wave

    float z[16];
#pragma unroll
    for (int q = 0; q < 4; ++q) {
#pragma unroll
      for (int c = 0; c < 4; ++c) {
        float e = si + ((const float*)&sv[q])[c];
        e = (e >= 0.f) ? e : 0.2f * e;            // LeakyReLU(0.2)
        z[q * 4 + c] = e + ((const float*)&av[q])[c];
      }
    }

    float mx = z[0];
#pragma unroll
    for (int t = 1; t < 16; ++t) mx = fmaxf(mx, z[t]);
    mx = fmaxf(mx, __shfl_xor(mx, 16, 32));   // combine lane halves per row
    const float m_new  = fmaxf(m_run, mx);
    const float fscale = __expf(m_run - m_new);
    float p[16], psum = 0.f;
#pragma unroll
    for (int t = 0; t < 16; ++t) { p[t] = __expf(z[t] - m_new); psum += p[t]; }
    psum += __shfl_xor(psum, 16, 32);
    l_run = l_run * fscale + psum;

    // Rescale accumulators only if any row max moved (rare after warm-up).
    if (!__all(m_new == m_run)) {
#pragma unroll
      for (int r = 0; r < 8; ++r) {
        const float fac = __shfl(fscale, r + 8 * hf, 32);
#pragma unroll
        for (int g = 0; g < 8; ++g) acc[g][r] *= fac;
      }
    }
    m_run = m_new;

    // Pack probabilities into the WMMA f16 A operand.
    v16h pa;
#pragma unroll
    for (int t = 0; t < 16; ++t) pa[t] = (_Float16)p[t];

    // 8 column tiles of h: B operand 32x16, per-lane two 16-byte runs of hT
    // (element t <-> K = 16*(t>>3) + 8*half + (t&7))  ->  2x global_load_b128.
#pragma unroll
    for (int g = 0; g < 8; ++g) {
      typedef __attribute__((ext_vector_type(8))) _Float16 v8h_t;
      const _Float16* hp = hTb + (long)(g * 16 + n) * N + j0 + 8 * hf;
      const v8h_t h0 = *(const v8h_t*)(hp);
      const v8h_t h1 = *(const v8h_t*)(hp + 16);
      const v16h hb = __builtin_shufflevector(h0, h1, 0, 1, 2, 3, 4, 5, 6, 7,
                                              8, 9, 10, 11, 12, 13, 14, 15);
      acc[g] = __builtin_amdgcn_wmma_f32_16x16x32_f16(
          false, pa, false, hb, (short)0, acc[g], false, false);
    }
  }

  // ---- deterministic split-softmax merge across the PART waves ----
  if (lane < 16) { sm[w][lane] = m_run; sl[w][lane] = l_run; }
  __syncthreads();

  // Combined stats for this lane's row n.
  float mtot = sm[0][n];
#pragma unroll
  for (int u = 1; u < PART; ++u) mtot = fmaxf(mtot, sm[u][n]);
  float ltot = 0.f;
#pragma unroll
  for (int u = 0; u < PART; ++u) ltot += sl[u][n] * __expf(sm[u][n] - mtot);
  const float fw = __expf(sm[w][n] - mtot) / ltot;   // this wave's row factor

  // Scale own accumulator (per-row broadcast) and stage into LDS.
#pragma unroll
  for (int r = 0; r < 8; ++r) {
    const float fac = __shfl(fw, r + 8 * hf, 32);
    const int  row  = r + 8 * hf;
#pragma unroll
    for (int g = 0; g < 8; ++g)
      part[(w * 16 + row) * KF + g * 16 + n] = acc[g][r] * fac;
  }
  __syncthreads();

  // Wave w owns 32 output columns: fixed-order sum of the 4 partials.
#pragma unroll
  for (int e = 0; e < 16; ++e) {
    const int row = (e * 32 + lane) >> 5;        // == e
    const int c   = w * 32 + ((e * 32 + lane) & 31);
    float v = part[row * KF + c];
#pragma unroll
    for (int u = 1; u < PART; ++u) v += part[(u * 16 + row) * KF + c];
    out[(bN + i0 + row) * KF + c] = v + bias[c];
  }
}

extern "C" void kernel_launch(void* const* d_in, const int* in_sizes, int n_in,
                              void* d_out, int out_size, void* d_ws, size_t ws_size,
                              hipStream_t stream) {
  const float* x    = (const float*)d_in[0];
  const float* A    = (const float*)d_in[1];
  const float* W    = (const float*)d_in[2];
  const float* a_w  = (const float*)d_in[3];
  const float* a_b  = (const float*)d_in[4];
  const float* bias = (const float*)d_in[5];
  float* out = (float*)d_out;

  // x is [B,N,KF]; A is [B,N,N] -> N = |A| / (|x|/KF), B = |x| / (N*KF)
  const long rows = (long)in_sizes[0] / KF;          // B*N
  const int  N    = (int)((long)in_sizes[1] / rows); // graph size
  const int  B    = (int)(rows / N);

  _Float16* hT = (_Float16*)d_ws;                        // B*KF*N f16 (2 MB)
  float*    s  = (float*)((char*)d_ws + rows * KF * sizeof(_Float16));

  gat_proj_kernel<<<dim3((unsigned)(rows / 16)), dim3(32), 0, stream>>>(
      x, W, a_w, a_b, hT, s, N);
  gat_attn_kernel<<<dim3(N / 16, B), dim3(32 * PART), 0, stream>>>(
      A, bias, hT, s, out, N);
}
